// LocalSpatialAttnLayer_45758581571631
// MI455X (gfx1250) — matrix operand order repaired and checked by reading, not compiled, not run on hardware
//
#include <hip/hip_runtime.h>
#include <hip/hip_bf16.h>
#include <math.h>

// ---------------------------------------------------------------------------
// LocalSpatialAttnLayer for MI455X (gfx1250, wave32, WMMA bf16 16x16x32)
//   B=32, N=1024 (32x32), C=128, NH=4, HD=32, MLP=512, conv K=3
// All GEMMs use v_wmma_f32_16x16x32_bf16 (f32 accumulate). All WMMA operands
// are fetched as contiguous 16B runs -> ds_load_b128 / global_load_b128.
// ---------------------------------------------------------------------------

typedef __bf16 bf16_t;
typedef __attribute__((ext_vector_type(16))) __bf16 v16bf;
typedef __attribute__((ext_vector_type(8)))  __bf16 v8bf;
typedef __attribute__((ext_vector_type(8)))  float  v8f;

#define DEV_INLINE __device__ __forceinline__

DEV_INLINE int lane_id() { return (int)(threadIdx.x & 31u); }

DEV_INLINE v8f vzero8() {
  v8f z = {0.f, 0.f, 0.f, 0.f, 0.f, 0.f, 0.f, 0.f};
  return z;
}

DEV_INLINE v8f wmma_bf16(v16bf a, v16bf b, v8f c) {
  return __builtin_amdgcn_wmma_f32_16x16x32_bf16(
      /*neg_a=*/false, a, /*neg_b=*/false, b,
      /*c_mod=*/(short)0, c, /*reuse_a=*/false, /*reuse_b=*/false);
}

DEV_INLINE v16bf combine16(v8bf lo, v8bf hi) {
  return __builtin_shufflevector(lo, hi, 0, 1, 2, 3, 4, 5, 6, 7,
                                 8, 9, 10, 11, 12, 13, 14, 15);
}

// A fragment: 16x32 (MxK) bf16 from row-major source elem(m,k) = p[m*ld+k].
// Lane (m=lane&15, kh=lane>>4) takes k = [kh*16, kh*16+15] (matches B)
//   ->  two 16B loads.
// Requires: p 16B-aligned and ld a multiple of 8 elements.
DEV_INLINE v16bf frag_a_row(const bf16_t* p, int ld) {
  const int lane = lane_id();
  const int m = lane & 15, kh = lane >> 4;
  const bf16_t* q = p + m * ld + (kh << 4);
  return combine16(*(const v8bf*)q, *(const v8bf*)(q + 8));
}

// B fragment: 32x16 (KxN) bf16 from N-major source elem(k,n) = p[n*ld+k].
// Lane (n=lane&15, kh=lane>>4) needs k = [kh*16, kh*16+15]: 32 contiguous B.
DEV_INLINE v16bf frag_b_nmaj(const bf16_t* p, int ld) {
  const int lane = lane_id();
  const int n = lane & 15, kh = lane >> 4;
  const bf16_t* q = p + n * ld + (kh << 4);
  return combine16(*(const v8bf*)q, *(const v8bf*)(q + 8));
}

// ---------------------------------------------------------------------------
// Weight prep (all B operands become N-major):
//   wt [tap][o][c] (9 x 384 x 128)  from Wqkv[o][c][kh][kw]
//   w1t[o][c]      (512 x 128)      = W1^T   (W1 is [128][512])
//   w2t[o][m]      (128 x 512)      = W2^T   (W2 is [512][128])
// ---------------------------------------------------------------------------
#define NQKVW (9 * 384 * 128)
#define NW1   (512 * 128)
#define NW2   (128 * 512)

__global__ void prep_weights_kernel(const float* __restrict__ Wqkv,
                                    const float* __restrict__ W1,
                                    const float* __restrict__ W2,
                                    bf16_t* __restrict__ wt,
                                    bf16_t* __restrict__ w1t,
                                    bf16_t* __restrict__ w2t) {
  const int i = blockIdx.x * 256 + threadIdx.x;
  if (i < NQKVW) {
    const int tap = i / (384 * 128);
    const int rem = i - tap * (384 * 128);
    const int o = rem >> 7;        // /128
    const int c = rem & 127;
    wt[i] = (bf16_t)Wqkv[(o * 128 + c) * 9 + tap];
  } else if (i < NQKVW + NW1) {
    const int j = i - NQKVW;
    const int o = j >> 7, c = j & 127;
    w1t[j] = (bf16_t)W1[c * 512 + o];
  } else if (i < NQKVW + NW1 + NW2) {
    const int j = i - NQKVW - NW1;
    const int o = j >> 9, m = j & 511;
    w2t[j] = (bf16_t)W2[m * 128 + o];
  }
}

// ---------------------------------------------------------------------------
// LayerNorm over C=128. One wave per row (32 lanes x 4 values).
// ---------------------------------------------------------------------------
__global__ void ln_kernel(const float* __restrict__ src,
                          const float* __restrict__ addsrc,
                          const float* __restrict__ g,
                          const float* __restrict__ b,
                          bf16_t* __restrict__ yout,
                          float* __restrict__ resout) {
  const int wid = threadIdx.x >> 5, lane = threadIdx.x & 31;
  const size_t row = (size_t)blockIdx.x * 8 + wid;
  const float* p = src + row * 128;

  float vals[4];
  float sum = 0.f;
#pragma unroll
  for (int i = 0; i < 4; ++i) {
    float t = p[lane + i * 32];
    if (addsrc) t += addsrc[row * 128 + lane + i * 32];
    vals[i] = t;
    sum += t;
  }
#pragma unroll
  for (int o = 16; o > 0; o >>= 1) sum += __shfl_xor(sum, o, 32);
  const float mu = sum * (1.f / 128.f);

  float vs = 0.f;
#pragma unroll
  for (int i = 0; i < 4; ++i) {
    const float d = vals[i] - mu;
    vs += d * d;
  }
#pragma unroll
  for (int o = 16; o > 0; o >>= 1) vs += __shfl_xor(vs, o, 32);
  const float rstd = rsqrtf(vs * (1.f / 128.f) + 1e-5f);

#pragma unroll
  for (int i = 0; i < 4; ++i) {
    const int c = lane + i * 32;
    const float yn = (vals[i] - mu) * rstd * g[c] + b[c];
    yout[row * 128 + c] = (bf16_t)yn;
    if (resout) resout[row * 128 + c] = vals[i];
  }
}

// ---------------------------------------------------------------------------
// QKV 3x3 conv as implicit GEMM: M=32 pixels/block, N=384, K=9*128.
// Weights are streamed as B fragments directly from global (N-major, cached
// in L2); only the shifted activation tile lives in LDS.
// Outputs: q,k in [B,NH,N,HD]; v transposed to [B,NH,HD,N] for the PV GEMM.
// Grid: 1024 blocks x 256 threads (8 waves), wave w owns cols [w*48,w*48+48).
// ---------------------------------------------------------------------------
#define ALD 40  // padded leading dim: 80B rows keep 16B alignment, break banks

__global__ void qkv_conv_kernel(const bf16_t* __restrict__ ln1,
                                const bf16_t* __restrict__ wt,
                                const float* __restrict__ bqkv,
                                bf16_t* __restrict__ qbuf,
                                bf16_t* __restrict__ kbuf,
                                bf16_t* __restrict__ vbuf) {
  __shared__ bf16_t As[32 * ALD];  // 32 pixels x 32 channels (padded)

  const int tid = threadIdx.x;
  const int wid = tid >> 5, lane = tid & 31;
  const int b = blockIdx.x >> 5;
  const int p0 = (blockIdx.x & 31) * 32;

  // Per-thread staging coords: thread copies 4 contiguous channels (8B).
  const int sr = tid >> 3;          // pixel row 0..31
  const int sc = (tid & 7) * 4;     // channel within 32-chunk
  const int spix = p0 + sr;
  const int sh = spix >> 5, sw = spix & 31;

  v8f acc[2][3];
#pragma unroll
  for (int i = 0; i < 2; ++i)
#pragma unroll
    for (int j = 0; j < 3; ++j) acc[i][j] = vzero8();

  for (int tap = 0; tap < 9; ++tap) {
    const int dh = tap / 3 - 1, dw = tap % 3 - 1;
    const int hh = sh + dh, ww = sw + dw;
    const bool inb = (hh >= 0 && hh < 32 && ww >= 0 && ww < 32);
    const bf16_t* srow = ln1 + ((size_t)b * 1024 + hh * 32 + ww) * 128;
    const bf16_t* wtap = wt + (size_t)tap * (384 * 128);
    __builtin_prefetch(wtap + (size_t)(wid * 48) * 128, 0, 0);

    for (int kc = 0; kc < 4; ++kc) {
      // Stage shifted activation chunk (zero outside image), b64 copies.
      uint2 val = {0u, 0u};
      if (inb) val = *(const uint2*)(srow + kc * 32 + sc);
      *(uint2*)(&As[sr * ALD + sc]) = val;
      __syncthreads();

      v16bf a0 = frag_a_row(&As[0], ALD);
      v16bf a1 = frag_a_row(&As[16 * ALD], ALD);
#pragma unroll
      for (int ct = 0; ct < 3; ++ct) {
        const bf16_t* bp = wtap + (size_t)(wid * 48 + ct * 16) * 128 + kc * 32;
        v16bf bf = frag_b_nmaj(bp, 128);
        acc[0][ct] = wmma_bf16(a0, bf, acc[0][ct]);
        acc[1][ct] = wmma_bf16(a1, bf, acc[1][ct]);
      }
      __syncthreads();
    }
  }

  // Epilogue: +bias, scatter to q/k ([B,NH,N,HD]) and vT ([B,NH,HD,N]).
  const int n = lane & 15, mh = lane >> 4;
#pragma unroll
  for (int rt = 0; rt < 2; ++rt) {
#pragma unroll
    for (int ct = 0; ct < 3; ++ct) {
      const int base = wid * 48 + ct * 16;   // col base (multiple of 16)
      const int which = base >> 7;           // 0=q 1=k 2=v (const per tile)
      const int head = (base >> 5) & 3;
      const int d = (base & 31) + n;
      const float bias = bqkv[base + n];
      const size_t bh = (size_t)b * 4 + head;
#pragma unroll
      for (int r = 0; r < 8; ++r) {
        const int pix = p0 + rt * 16 + mh * 8 + r;
        const float val = acc[rt][ct][r] + bias;
        if (which == 2) {
          vbuf[(bh * 32 + d) * 1024 + pix] = (bf16_t)val;           // vT
        } else {
          bf16_t* dst = which ? kbuf : qbuf;
          dst[(bh * 1024 + pix) * 32 + d] = (bf16_t)val;
        }
      }
    }
  }
}

// ---------------------------------------------------------------------------
// Attention: one (b, head, 32-query block) per workgroup.
// Dynamic LDS: S[32][1024] f32 + Pb[32][1032] bf16 + Q tile + O tile.
// Phase 1: S = scale * Q K^T  (HD=32 -> one WMMA per 16x16 tile)
// Phase 2: exact softmax; normalized probs written once as bf16 (Pb)
// Phase 3: O = P V, split-K across wave pairs, ds_add_f32 combine.
// ---------------------------------------------------------------------------
#define QLD 40
#define PLD 1032
#define ATTN_S_BYTES  (32 * 1024 * 4)
#define ATTN_P_BYTES  (32 * PLD * 2)
#define ATTN_Q_BYTES  (32 * QLD * 2)
#define ATTN_LDS (ATTN_S_BYTES + ATTN_P_BYTES + ATTN_Q_BYTES + 32 * 32 * 4)

__global__ void attn_kernel(const bf16_t* __restrict__ q,
                            const bf16_t* __restrict__ k,
                            const bf16_t* __restrict__ vt,
                            float* __restrict__ attn_out) {
  extern __shared__ char smem_raw[];
  float*  S   = (float*)smem_raw;
  bf16_t* Pb  = (bf16_t*)(smem_raw + ATTN_S_BYTES);
  bf16_t* Qs  = (bf16_t*)(smem_raw + ATTN_S_BYTES + ATTN_P_BYTES);
  float*  Osh = (float*)(smem_raw + ATTN_S_BYTES + ATTN_P_BYTES + ATTN_Q_BYTES);

  const int tid = threadIdx.x;
  const int wid = tid >> 5, lane = tid & 31;
  const int bh = blockIdx.x >> 5;               // b*4 + head
  const int q0 = (blockIdx.x & 31) * 32;
  const int b = bh >> 2, head = bh & 3;

  const bf16_t* qb = q + (size_t)bh * 1024 * 32;
  const bf16_t* kb = k + (size_t)bh * 1024 * 32;
  const bf16_t* vb = vt + (size_t)bh * 32 * 1024;

  // Stage Q tile (b128 copies; LDS rows padded to QLD) + zero O tile.
  if (tid < 128) {
    const int r = tid >> 2, c = tid & 3;  // 4 x 16B per 64B row
    *(uint4*)(&Qs[r * QLD + c * 8]) =
        *(const uint4*)(qb + (size_t)(q0 + r) * 32 + c * 8);
  }
  for (int i = tid; i < 32 * 32; i += 256) Osh[i] = 0.f;
  __syncthreads();

  // ---- Phase 1: scores
  const float scale = 0.17677669529663687f;  // 1/sqrt(32)
  const int n = lane & 15, mh = lane >> 4;
  v16bf aq0 = frag_a_row(Qs, QLD);
  v16bf aq1 = frag_a_row(Qs + 16 * QLD, QLD);
  for (int ct = 0; ct < 8; ++ct) {
    const int c0 = wid * 128 + ct * 16;            // key columns
    v16bf bf = frag_b_nmaj(kb + (size_t)c0 * 32, 32);  // B(k=d, n=key)
    v8f s0 = wmma_bf16(aq0, bf, vzero8());
    v8f s1 = wmma_bf16(aq1, bf, vzero8());
#pragma unroll
    for (int r = 0; r < 8; ++r) {
      S[(mh * 8 + r) * 1024 + c0 + n]      = s0[r] * scale;
      S[(16 + mh * 8 + r) * 1024 + c0 + n] = s1[r] * scale;
    }
  }
  __syncthreads();

  // ---- Phase 2: softmax (each wave owns 4 rows); emit bf16 probs to Pb.
  for (int rr = 0; rr < 4; ++rr) {
    const int row = wid * 4 + rr;
    float* Sr = S + row * 1024;
    bf16_t* Pr = Pb + row * PLD;
    float mx = -3.0e38f;
    for (int c = lane; c < 1024; c += 32) mx = fmaxf(mx, Sr[c]);
#pragma unroll
    for (int o = 16; o > 0; o >>= 1) mx = fmaxf(mx, __shfl_xor(mx, o, 32));
    float sum = 0.f;
    for (int c = lane; c < 1024; c += 32) {
      const float e = __expf(Sr[c] - mx);
      Sr[c] = e;
      sum += e;
    }
#pragma unroll
    for (int o = 16; o > 0; o >>= 1) sum += __shfl_xor(sum, o, 32);
    const float inv = 1.f / sum;
    for (int c = lane; c < 1024; c += 32) Pr[c] = (bf16_t)(Sr[c] * inv);
  }
  __syncthreads();

  // ---- Phase 3: O = P V  (4 output tiles, split-K over wave pairs)
  {
    const int t = wid & 3;
    const int rt = t >> 1, ctile = t & 1;
    const int kbase = (wid >> 2) * 512;
    v8f acc = vzero8();
    for (int kc = 0; kc < 16; ++kc) {
      const int k0 = kbase + kc * 32;
      v16bf a = frag_a_row(Pb + (size_t)(rt * 16) * PLD + k0, PLD);
      v16bf bf = frag_b_nmaj(vb + (size_t)(ctile * 16) * 1024 + k0, 1024);
      acc = wmma_bf16(a, bf, acc);
    }
#pragma unroll
    for (int r = 0; r < 8; ++r)
      atomicAdd(&Osh[(rt * 16 + mh * 8 + r) * 32 + ctile * 16 + n], acc[r]);
  }
  __syncthreads();

  for (int i = tid; i < 32 * 32; i += 256) {
    const int r = i >> 5, d = i & 31;
    attn_out[((size_t)b * 1024 + q0 + r) * 128 + head * 32 + d] = Osh[i];
  }
}

// ---------------------------------------------------------------------------
// MLP fc1 + exact GELU: [32768,128] x [128,512] -> bf16 [32768,512]
// Grid: 1024 blocks x 256 threads; wave w owns cols [w*64, w*64+64).
// ---------------------------------------------------------------------------
#define M1LD 136  // 272B rows: 16B-aligned, bank-conflict friendly

__global__ void mlp1_kernel(const bf16_t* __restrict__ y2,
                            const bf16_t* __restrict__ w1t,
                            const float* __restrict__ bf1,
                            bf16_t* __restrict__ m1) {
  __shared__ bf16_t As[32 * M1LD];
  const int tid = threadIdx.x;
  const int wid = tid >> 5, lane = tid & 31;
  const int r0 = blockIdx.x * 32;

  // Stage 32x128 activation tile as b128 copies (16 x 16B per row).
  for (int i = tid; i < 512; i += 256) {
    const int r = i >> 4, c = i & 15;
    *(uint4*)(&As[r * M1LD + c * 8]) =
        *(const uint4*)(y2 + (size_t)(r0 + r) * 128 + c * 8);
  }
  __syncthreads();

  v8f acc[2][4];
#pragma unroll
  for (int i = 0; i < 2; ++i)
#pragma unroll
    for (int j = 0; j < 4; ++j) acc[i][j] = vzero8();

  const int col0 = wid * 64;
  __builtin_prefetch(w1t + (size_t)col0 * 128, 0, 0);
  for (int kc = 0; kc < 4; ++kc) {
    v16bf a0 = frag_a_row(&As[kc * 32], M1LD);
    v16bf a1 = frag_a_row(&As[16 * M1LD + kc * 32], M1LD);
#pragma unroll
    for (int ct = 0; ct < 4; ++ct) {
      const bf16_t* bp = w1t + (size_t)(col0 + ct * 16) * 128 + kc * 32;
      v16bf bf = frag_b_nmaj(bp, 128);
      acc[0][ct] = wmma_bf16(a0, bf, acc[0][ct]);
      acc[1][ct] = wmma_bf16(a1, bf, acc[1][ct]);
    }
  }

  const int n = lane & 15, mh = lane >> 4;
#pragma unroll
  for (int rt = 0; rt < 2; ++rt) {
#pragma unroll
    for (int ct = 0; ct < 4; ++ct) {
      const int o = col0 + ct * 16 + n;
      const float bias = bf1[o];
#pragma unroll
      for (int r = 0; r < 8; ++r) {
        const int row = r0 + rt * 16 + mh * 8 + r;
        const float x = acc[rt][ct][r] + bias;
        const float gelu = 0.5f * x * (1.f + erff(x * 0.7071067811865475f));
        m1[(size_t)row * 512 + o] = (bf16_t)gelu;
      }
    }
  }
}

// ---------------------------------------------------------------------------
// MLP fc2 + bias + residual: [32768,512] x [512,128] -> f32 out
// Grid: 1024 blocks x 256 threads; wave w owns col tile [w*16, w*16+16).
// ---------------------------------------------------------------------------
#define M2LD 520  // 1040B rows: 16B-aligned

__global__ void mlp2_kernel(const bf16_t* __restrict__ m1,
                            const bf16_t* __restrict__ w2t,
                            const float* __restrict__ bf2,
                            const float* __restrict__ res2,
                            float* __restrict__ out) {
  __shared__ bf16_t As[32 * M2LD];
  const int tid = threadIdx.x;
  const int wid = tid >> 5, lane = tid & 31;
  const int r0 = blockIdx.x * 32;

  // Stage 32x512 activation tile as b128 copies (64 x 16B per row).
  for (int i = tid; i < 2048; i += 256) {
    const int r = i >> 6, c = i & 63;
    *(uint4*)(&As[r * M2LD + c * 8]) =
        *(const uint4*)(m1 + (size_t)(r0 + r) * 512 + c * 8);
  }
  __syncthreads();

  v8f acc0 = vzero8(), acc1 = vzero8();
  const int col0 = wid * 16;
  __builtin_prefetch(w2t + (size_t)col0 * 512, 0, 0);
  for (int kc = 0; kc < 16; ++kc) {
    v16bf a0 = frag_a_row(&As[kc * 32], M2LD);
    v16bf a1 = frag_a_row(&As[16 * M2LD + kc * 32], M2LD);
    v16bf bf = frag_b_nmaj(w2t + (size_t)col0 * 512 + kc * 32, 512);
    acc0 = wmma_bf16(a0, bf, acc0);
    acc1 = wmma_bf16(a1, bf, acc1);
  }

  const int n = lane & 15, mh = lane >> 4;
  const int o = col0 + n;
  const float bias = bf2[o];
#pragma unroll
  for (int r = 0; r < 8; ++r) {
    const size_t row0 = (size_t)(r0 + mh * 8 + r) * 128;
    const size_t row1 = (size_t)(r0 + 16 + mh * 8 + r) * 128;
    out[row0 + o] = acc0[r] + bias + res2[row0 + o];
    out[row1 + o] = acc1[r] + bias + res2[row1 + o];
  }
}

// ---------------------------------------------------------------------------
// Host launch
// ---------------------------------------------------------------------------
extern "C" void kernel_launch(void* const* d_in, const int* in_sizes, int n_in,
                              void* d_out, int out_size, void* d_ws, size_t ws_size,
                              hipStream_t stream) {
  (void)in_sizes; (void)n_in; (void)out_size; (void)ws_size;

  const float* x    = (const float*)d_in[0];
  const float* Wqkv = (const float*)d_in[1];
  const float* bqkv = (const float*)d_in[2];
  const float* g1   = (const float*)d_in[3];
  const float* b1   = (const float*)d_in[4];
  const float* g2   = (const float*)d_in[5];
  const float* b2   = (const float*)d_in[6];
  const float* W1   = (const float*)d_in[7];
  const float* bf1  = (const float*)d_in[8];
  const float* W2   = (const float*)d_in[9];
  const float* bf2  = (const float*)d_in[10];

  char* ws = (char*)d_ws;
  size_t off = 0;
  auto take = [&](size_t bytes) -> char* {
    char* p = ws + off;
    off = (off + bytes + 255) & ~(size_t)255;
    return p;
  };

  const size_t BN = 32u * 1024u;  // 32768 rows
  bf16_t* wt   = (bf16_t*)take((size_t)NQKVW * 2);
  bf16_t* w1t  = (bf16_t*)take((size_t)NW1 * 2);
  bf16_t* w2t  = (bf16_t*)take((size_t)NW2 * 2);
  bf16_t* ln1  = (bf16_t*)take(BN * 128 * 2);   // reused as y2 after conv
  bf16_t* qb   = (bf16_t*)take(BN * 128 * 2);
  bf16_t* kb   = (bf16_t*)take(BN * 128 * 2);
  bf16_t* vtb  = (bf16_t*)take(BN * 128 * 2);   // v transposed [B,NH,HD,N]
  float*  attn = (float*)take(BN * 128 * 4);
  float*  res2 = (float*)take(BN * 128 * 4);
  bf16_t* m1   = (bf16_t*)take(BN * 512 * 2);

  (void)hipFuncSetAttribute((const void*)attn_kernel,
                            hipFuncAttributeMaxDynamicSharedMemorySize,
                            ATTN_LDS);

  const int nprep = NQKVW + NW1 + NW2;
  prep_weights_kernel<<<(nprep + 255) / 256, 256, 0, stream>>>(Wqkv, W1, W2,
                                                               wt, w1t, w2t);
  ln_kernel<<<4096, 256, 0, stream>>>(x, nullptr, g1, b1, ln1, nullptr);
  qkv_conv_kernel<<<1024, 256, 0, stream>>>(ln1, wt, bqkv, qb, kb, vtb);
  attn_kernel<<<4096, 256, ATTN_LDS, stream>>>(qb, kb, vtb, attn);
  ln_kernel<<<4096, 256, 0, stream>>>(attn, x, g2, b2, ln1 /*=y2*/, res2);
  mlp1_kernel<<<1024, 256, 0, stream>>>(ln1, w1t, bf1, m1);
  mlp2_kernel<<<1024, 256, 0, stream>>>(m1, w2t, bf2, res2, (float*)d_out);
}
